// WorldModelRNN_16535624089722
// MI455X (gfx1250) — matrix operand order repaired and checked by reading, not compiled, not run on hardware
//
#include <hip/hip_runtime.h>

#define BATCH   1024
#define TT      128
#define NROLL   64
#define HIST    (TT - NROLL)     // 64
#define RSTEPS  (NROLL - 1)      // 63
#define HDIM    256
#define NLAYER  8
#define OUTDIM  9

typedef __attribute__((ext_vector_type(16))) __bf16 v16bf;
typedef __attribute__((ext_vector_type(8)))  float  v8f;

struct alignas(16) U128 { unsigned int a, b, c, d; };
union FragU { v16bf v; U128 q[2]; };

__device__ __forceinline__ unsigned short f2bf(float f) {
    unsigned int u = __float_as_uint(f);
    u += 0x7FFFu + ((u >> 16) & 1u);          // round-to-nearest-even
    return (unsigned short)(u >> 16);
}

__device__ __forceinline__ float tanh_fast(float x) {
    x = fminf(fmaxf(x, -15.0f), 15.0f);
    float e = __builtin_amdgcn_exp2f(x * 2.8853900817779268f); // e^(2x)
    return (e - 1.0f) * __builtin_amdgcn_rcpf(e + 1.0f);
}

// A fragment: 16x32 bf16, row-major source with given stride (elements).
// lane&15 = row M; lanes 0-15 take K[0..8)+K[16..24), lanes 16-31 K[8..16)+K[24..32).
__device__ __forceinline__ v16bf ld_a(const unsigned short* base, int stride, int lane) {
    const int row  = lane & 15;
    const int koff = (lane >> 4) << 3;
    const unsigned short* p = base + row * stride + koff;
    FragU u;
    u.q[0] = *(const U128*)(p);
    u.q[1] = *(const U128*)(p + 16);
    return u.v;
}

// B fragment: 32x16 bf16 where B[k][n] = W[n][k]; W row-major with given stride.
// lane&15 = col N; lanes 0-15 take K[0..8)+K[16..24), lanes 16-31 K[8..16)+K[24..32).
__device__ __forceinline__ v16bf ld_b(const unsigned short* base, int stride, int lane) {
    const int n  = lane & 15;
    const int kb = (lane >> 4) << 3;
    const unsigned short* p = base + n * stride + kb;
    FragU u;
    u.q[0] = *(const U128*)(p);
    u.q[1] = *(const U128*)(p + 16);
    return u.v;
}

__device__ __forceinline__ v8f wmma_bf16(v16bf a, v16bf b, v8f c) {
    return __builtin_amdgcn_wmma_f32_16x16x32_bf16(false, a, false, b, (short)0, c, false, false);
}

// One RNN layer for a 16-row batch tile: hnew = tanh(x @ Wih^T + hold @ Whh^T + bias)
// Each wave computes 2 N-tiles of 16 columns. Caller barriers afterwards.
__device__ __forceinline__ void rnn_layer(
    const unsigned short* __restrict__ xin, int xstride, int xkt,
    const unsigned short* __restrict__ hold,
    unsigned short*       __restrict__ hnew,
    const unsigned short* __restrict__ Wih, int wihStride,
    const unsigned short* __restrict__ Whh,
    const float*          __restrict__ bias,
    int wave, int lane)
{
    const int nt0 = wave * 2, nt1 = nt0 + 1;
    v8f acc0 = {};
    v8f acc1 = {};
    const unsigned short* w0 = Wih + nt0 * 16 * wihStride;
    const unsigned short* w1 = Wih + nt1 * 16 * wihStride;
#pragma unroll
    for (int kt = 0; kt < xkt; ++kt) {
        v16bf a = ld_a(xin + kt * 32, xstride, lane);
        acc0 = wmma_bf16(a, ld_b(w0 + kt * 32, wihStride, lane), acc0);
        acc1 = wmma_bf16(a, ld_b(w1 + kt * 32, wihStride, lane), acc1);
    }
    const unsigned short* v0 = Whh + nt0 * 16 * HDIM;
    const unsigned short* v1 = Whh + nt1 * 16 * HDIM;
#pragma unroll
    for (int kt = 0; kt < 8; ++kt) {
        v16bf a = ld_a(hold + kt * 32, HDIM, lane);
        acc0 = wmma_bf16(a, ld_b(v0 + kt * 32, HDIM, lane), acc0);
        acc1 = wmma_bf16(a, ld_b(v1 + kt * 32, HDIM, lane), acc1);
    }
    const int col = lane & 15;
    const int mb  = (lane >> 4) * 8;
    const float b0v = bias[nt0 * 16 + col];
    const float b1v = bias[nt1 * 16 + col];
#pragma unroll
    for (int r = 0; r < 8; ++r) {
        const int m = mb + r;
        hnew[m * HDIM + nt0 * 16 + col] = f2bf(tanh_fast(acc0[r] + b0v));
        hnew[m * HDIM + nt1 * 16 + col] = f2bf(tanh_fast(acc1[r] + b1v));
    }
}

// 8-layer stacked step; p = parity of "old" hidden buffers (reads buf[p], writes buf[1-p]).
__device__ __forceinline__ void rnn_step(
    const unsigned short* x0, unsigned short* hbuf, int p,
    const unsigned short* Wih0p, const unsigned short* Wih, const unsigned short* Whh,
    const float* bias, int wave, int lane)
{
    const int TILE = 16 * HDIM;
    rnn_layer(x0, 32, 1,
              hbuf + (p * NLAYER + 0) * TILE,
              hbuf + ((1 - p) * NLAYER + 0) * TILE,
              Wih0p, 32, Whh, bias, wave, lane);
    __syncthreads();
#pragma unroll 1
    for (int l = 1; l < NLAYER; ++l) {
        rnn_layer(hbuf + ((1 - p) * NLAYER + (l - 1)) * TILE, HDIM, 8,
                  hbuf + (p * NLAYER + l) * TILE,
                  hbuf + ((1 - p) * NLAYER + l) * TILE,
                  Wih + (l - 1) * HDIM * HDIM, HDIM,
                  Whh + l * HDIM * HDIM,
                  bias + l * HDIM, wave, lane);
        __syncthreads();
    }
}

__global__ __launch_bounds__(256) void wm_rnn_main(
    const float* __restrict__ dts, const float* __restrict__ states, const float* __restrict__ acts,
    const unsigned short* __restrict__ encWih0p, const unsigned short* __restrict__ encWih,
    const unsigned short* __restrict__ encWhh,   const float* __restrict__ encBias,
    const unsigned short* __restrict__ rnnWih0p, const unsigned short* __restrict__ rnnWih,
    const unsigned short* __restrict__ rnnWhh,   const float* __restrict__ rnnBias,
    const unsigned short* __restrict__ fcWp,     const float* __restrict__ fcb,
    float* __restrict__ out)
{
    extern __shared__ unsigned short smem[];
    unsigned short* hbuf = smem;                                 // 2*8*16*256 bf16 = 128KB
    unsigned short* x0   = smem + 2 * NLAYER * 16 * HDIM;        // 16*32 bf16 (K-padded input)
    float*          pred = (float*)(x0 + 16 * 32);               // 16x16 f32 feedback

    const int tid  = threadIdx.x;
    const int wave = tid >> 5, lane = tid & 31;
    const int b0   = blockIdx.x * 16;

    for (int i = tid; i < 2 * NLAYER * 16 * HDIM; i += 256) hbuf[i] = 0;
    __syncthreads();

    int p = 0;

    // ---------------- encoder: 64 steps ----------------
#pragma unroll 1
    for (int t = 0; t < HIST; ++t) {
        if (tid < 16) {
            const int b = b0 + tid;
            unsigned short* xr = x0 + tid * 32;
            xr[0] = f2bf(dts[b * TT + t + 1]);
#pragma unroll
            for (int j = 0; j < 9; ++j) xr[1 + j] = f2bf(states[(b * TT + t) * 12 + 3 + j]);
#pragma unroll
            for (int j = 0; j < 4; ++j) xr[10 + j] = f2bf(acts[(b * TT + t) * 4 + j]);
#pragma unroll
            for (int j = 14; j < 32; ++j) xr[j] = 0;
        }
        __syncthreads();
        rnn_step(x0, hbuf, p, encWih0p, encWih, encWhh, encBias, wave, lane);
        p ^= 1;
    }

    // pred0 = states[:, HIST, 3:12]
    if (tid < 16) {
        const int b = b0 + tid;
#pragma unroll
        for (int j = 0; j < OUTDIM; ++j) pred[tid * 16 + j] = states[(b * TT + HIST) * 12 + 3 + j];
    }
    __syncthreads();

    // ---------------- rollout: 63 autoregressive steps ----------------
#pragma unroll 1
    for (int r = 0; r < RSTEPS; ++r) {
        if (tid < 16) {
            const int b = b0 + tid;
            unsigned short* xr = x0 + tid * 32;
            xr[0] = f2bf(dts[b * TT + HIST + 1 + r]);
#pragma unroll
            for (int j = 0; j < OUTDIM; ++j) xr[1 + j] = f2bf(pred[tid * 16 + j]);
#pragma unroll
            for (int j = 0; j < 4; ++j) xr[10 + j] = f2bf(acts[(b * TT + HIST + r) * 4 + j]);
#pragma unroll
            for (int j = 14; j < 32; ++j) xr[j] = 0;
        }
        __syncthreads();
        rnn_step(x0, hbuf, p, rnnWih0p, rnnWih, rnnWhh, rnnBias, wave, lane);
        const unsigned short* top = hbuf + ((1 - p) * NLAYER + 7) * 16 * HDIM;
        p ^= 1;

        // FC: out = top @ fcW^T + fcb  (16x9, padded to one 16x16 WMMA tile on wave 0)
        if (wave == 0) {
            v8f acc = {};
#pragma unroll
            for (int kt = 0; kt < 8; ++kt) {
                v16bf a = ld_a(top + kt * 32, HDIM, lane);
                acc = wmma_bf16(a, ld_b(fcWp + kt * 32, HDIM, lane), acc);
            }
            const int n = lane & 15, mb = (lane >> 4) * 8;
            if (n < OUTDIM) {
                const float bb = fcb[n];
#pragma unroll
                for (int rr = 0; rr < 8; ++rr) {
                    const int m = mb + rr;
                    const float v = acc[rr] + bb;
                    pred[m * 16 + n] = v;
                    out[((long)(b0 + m) * RSTEPS + r) * OUTDIM + n] = v;
                }
            }
        }
        __syncthreads();
    }
}

// ---------------- weight pre-pass kernels ----------------
__global__ void cvt_bf16(const float* __restrict__ src, unsigned short* __restrict__ dst, int n) {
    int i = blockIdx.x * blockDim.x + threadIdx.x;
    if (i < n) dst[i] = f2bf(src[i]);
}

__global__ void cvt_pad(const float* __restrict__ src, unsigned short* __restrict__ dst,
                        int srows, int scols, int drows, int dcols) {
    int i = blockIdx.x * blockDim.x + threadIdx.x;
    if (i >= drows * dcols) return;
    int r = i / dcols, c = i % dcols;
    dst[i] = (r < srows && c < scols) ? f2bf(src[r * scols + c]) : (unsigned short)0;
}

__global__ void bias_sum(const float* __restrict__ a, const float* __restrict__ b,
                         float* __restrict__ dst, int n) {
    int i = blockIdx.x * blockDim.x + threadIdx.x;
    if (i < n) dst[i] = a[i] + b[i];
}

extern "C" void kernel_launch(void* const* d_in, const int* in_sizes, int n_in,
                              void* d_out, int out_size, void* d_ws, size_t ws_size,
                              hipStream_t stream)
{
    const float* dts     = (const float*)d_in[0];
    const float* states  = (const float*)d_in[1];
    const float* acts    = (const float*)d_in[2];
    // d_in[3] = num_rollout (compile-time constant NROLL here)
    const float* encWih0 = (const float*)d_in[4];
    const float* encWihF = (const float*)d_in[5];
    const float* encWhhF = (const float*)d_in[6];
    const float* encBih  = (const float*)d_in[7];
    const float* encBhh  = (const float*)d_in[8];
    const float* rnnWih0 = (const float*)d_in[9];
    const float* rnnWihF = (const float*)d_in[10];
    const float* rnnWhhF = (const float*)d_in[11];
    const float* rnnBih  = (const float*)d_in[12];
    const float* rnnBhh  = (const float*)d_in[13];
    const float* fcW     = (const float*)d_in[14];
    const float* fcb     = (const float*)d_in[15];

    unsigned char* ws = (unsigned char*)d_ws;
    unsigned short* encWih0p = (unsigned short*)(ws + 0);        // 256x32 bf16 (K-padded)
    unsigned short* encWihB  = (unsigned short*)(ws + 16384);    // 7x256x256 bf16
    unsigned short* encWhhB  = (unsigned short*)(ws + 933888);   // 8x256x256 bf16
    unsigned short* rnnWih0p = (unsigned short*)(ws + 1982464);
    unsigned short* rnnWihB  = (unsigned short*)(ws + 1998848);
    unsigned short* rnnWhhB  = (unsigned short*)(ws + 2916352);
    unsigned short* fcWp     = (unsigned short*)(ws + 3964928);  // 16x256 bf16 (M-padded)
    float*          encBias  = (float*)(ws + 3973120);           // 8x256 f32 (bih+bhh)
    float*          rnnBias  = (float*)(ws + 3981312);

    auto g = [](int n) { return (n + 255) / 256; };
    cvt_pad <<<g(256 * 32),  256, 0, stream>>>(encWih0, encWih0p, 256, 14, 256, 32);
    cvt_bf16<<<g(7 * 65536), 256, 0, stream>>>(encWihF, encWihB, 7 * 65536);
    cvt_bf16<<<g(8 * 65536), 256, 0, stream>>>(encWhhF, encWhhB, 8 * 65536);
    cvt_pad <<<g(256 * 32),  256, 0, stream>>>(rnnWih0, rnnWih0p, 256, 14, 256, 32);
    cvt_bf16<<<g(7 * 65536), 256, 0, stream>>>(rnnWihF, rnnWihB, 7 * 65536);
    cvt_bf16<<<g(8 * 65536), 256, 0, stream>>>(rnnWhhF, rnnWhhB, 8 * 65536);
    cvt_pad <<<g(16 * 256),  256, 0, stream>>>(fcW, fcWp, 9, 256, 16, 256);
    bias_sum<<<g(2048),      256, 0, stream>>>(encBih, encBhh, encBias, 2048);
    bias_sum<<<g(2048),      256, 0, stream>>>(rnnBih, rnnBhh, rnnBias, 2048);

    const size_t shmem = (size_t)(2 * NLAYER * 16 * HDIM + 16 * 32) * sizeof(unsigned short)
                       + (size_t)(16 * 16) * sizeof(float);     // ~130 KB, fits 320 KB/WGP
    wm_rnn_main<<<BATCH / 16, 256, shmem, stream>>>(
        dts, states, acts,
        encWih0p, encWihB, encWhhB, encBias,
        rnnWih0p, rnnWihB, rnnWhhB, rnnBias,
        fcWp, fcb, (float*)d_out);
}
